// MaskedSelfAttention_498216206782
// MI455X (gfx1250) — matrix operand order repaired and checked by reading, not compiled, not run on hardware
//
#include <hip/hip_runtime.h>
#include <cmath>

typedef __attribute__((ext_vector_type(16))) __bf16 v16bf;
typedef __attribute__((ext_vector_type(4)))  __bf16 v4bf;
typedef __attribute__((ext_vector_type(8)))  float  v8f;
typedef __attribute__((ext_vector_type(4)))  int    v4i;

#define T_SEQ 2048
#define DHEAD 128
#define NHEAD 16
#define NGRP  4
#define WIN   512
#define SINK  4

#define GLOBAL_AS __attribute__((address_space(1)))
#define LDS_AS    __attribute__((address_space(3)))

static __device__ __forceinline__ int imin(int a, int b) { return a < b ? a : b; }

union Frag {
  v16bf v;
  unsigned u[8];
  __bf16 h[16];
};

__device__ __forceinline__ v8f wmma_bf16(v16bf a, v16bf b, v8f c) {
  // D = A(16x32 bf16) * B(32x16 bf16) + C(16x16 f32)
  return __builtin_amdgcn_wmma_f32_16x16x32_bf16(false, a, false, b, (short)0, c,
                                                 false, false);
}

// fp32 -> bf16 hi + residual lo (fp32-accuracy emulation on the bf16 WMMA pipe)
__device__ __forceinline__ void split_pair(float x, float y, v16bf& hi, v16bf& lo,
                                           int e) {
  __bf16 xh = (__bf16)x, yh = (__bf16)y;
  hi[e]     = xh;
  hi[e + 1] = yh;
  lo[e]     = (__bf16)(x - (float)xh);
  lo[e + 1] = (__bf16)(y - (float)yh);
}

__device__ __forceinline__ void cvt4(const float4 f, v4bf& hi, v4bf& lo) {
  __bf16 h0 = (__bf16)f.x, h1 = (__bf16)f.y, h2 = (__bf16)f.z, h3 = (__bf16)f.w;
  hi[0] = h0; hi[1] = h1; hi[2] = h2; hi[3] = h3;
  lo[0] = (__bf16)(f.x - (float)h0);
  lo[1] = (__bf16)(f.y - (float)h1);
  lo[2] = (__bf16)(f.z - (float)h2);
  lo[3] = (__bf16)(f.w - (float)h3);
}

// Async global->LDS 16B copy (ASYNCcnt path) with safe fallback.
__device__ __forceinline__ void async_copy16(const float* g, float* l) {
#if __has_builtin(__builtin_amdgcn_global_load_async_to_lds_b128)
  __builtin_amdgcn_global_load_async_to_lds_b128(
      (GLOBAL_AS v4i*)((GLOBAL_AS void*)g), (LDS_AS v4i*)((LDS_AS void*)l), 0, 0);
#else
  *(float4*)l = *(const float4*)g;
#endif
}

__device__ __forceinline__ void async_wait0() {
#if __has_builtin(__builtin_amdgcn_s_wait_asynccnt)
  __builtin_amdgcn_s_wait_asynccnt(0);
#elif __has_builtin(__builtin_amdgcn_global_load_async_to_lds_b128)
  asm volatile("s_wait_asynccnt 0" ::: "memory");
#endif
}

// ---------------------------------------------------------------------------
// Out[M,N] = A[M,K] @ W[K,N] + bias  (fp32 io, split-bf16 WMMA internally)
// Block: 256 thr = 8 waves (4x2), wave tile 16x32, block tile 64x64, BK=32.
// Tiles staged async into LDS as f32, split-converted ONCE into bf16 hi/lo.
// ---------------------------------------------------------------------------
__global__ void __launch_bounds__(256)
gemm_bias_kernel(const float* __restrict__ A, const float* __restrict__ W,
                 const float* __restrict__ bias, float* __restrict__ Out,
                 int M, int N, int Kd) {
  __shared__ __align__(16) float rawA[64 * 32];
  __shared__ __align__(16) float rawB[32 * 64];
  __shared__ __align__(16) __bf16 sAhi[64][40];  // [m][k]
  __shared__ __align__(16) __bf16 sAlo[64][40];
  __shared__ __align__(16) __bf16 sBhi[64][40];  // transposed [n][k]
  __shared__ __align__(16) __bf16 sBlo[64][40];
  const int tid  = threadIdx.x;
  const int wave = tid >> 5, lane = tid & 31;
  const int wm = wave >> 1, wn = wave & 1;
  const int half = lane >> 4, ln = lane & 15;
  const size_t m0 = (size_t)blockIdx.x * 64;
  const size_t n0 = (size_t)blockIdx.y * 64;

  const v8f zero8 = {0.f, 0.f, 0.f, 0.f, 0.f, 0.f, 0.f, 0.f};
  v8f c0 = zero8, c1 = zero8;

  for (int k0 = 0; k0 < Kd; k0 += 32) {
    // ---- stage raw f32 tiles (async global->LDS, 2+2 x b128 per thread) ----
#pragma unroll
    for (int i = 0; i < 2; i++) {
      const int idx = tid + i * 256;
      {
        const int r = idx >> 3, c4 = (idx & 7) * 4;  // A: 64x32
        async_copy16(A + (m0 + r) * (size_t)Kd + k0 + c4, &rawA[idx * 4]);
      }
      {
        const int r = idx >> 4, c4 = (idx & 15) * 4;  // W: 32x64
        async_copy16(W + (size_t)(k0 + r) * N + n0 + c4, &rawB[idx * 4]);
      }
    }
    async_wait0();
    __syncthreads();
    // ---- split-convert once per element into bf16 hi/lo tiles ----
    {
      const int r = tid >> 2, cA = (tid & 3) * 8;  // A: 8 elems/thread
#pragma unroll
      for (int j = 0; j < 2; j++) {
        float4 f = *(const float4*)&rawA[r * 32 + cA + j * 4];
        v4bf h, l;
        cvt4(f, h, l);
        *(v4bf*)&sAhi[r][cA + j * 4] = h;
        *(v4bf*)&sAlo[r][cA + j * 4] = l;
      }
      const int rb = tid >> 3, cB = (tid & 7) * 8;  // B: 8 elems, transposed
#pragma unroll
      for (int j = 0; j < 2; j++) {
        float4 f = *(const float4*)&rawB[rb * 64 + cB + j * 4];
        v4bf h, l;
        cvt4(f, h, l);
#pragma unroll
        for (int e = 0; e < 4; e++) {
          sBhi[cB + j * 4 + e][rb] = h[e];
          sBlo[cB + j * 4 + e][rb] = l[e];
        }
      }
    }
    __syncthreads();
    // ---- packed fragment loads + 6 WMMAs ----
    Frag ah, al, bh0, bl0, bh1, bl1;
    const int arow = wm * 16 + ln;
    const int bcol = wn * 32 + ln;
#pragma unroll
    for (int p = 0; p < 8; p++) {
      const int ka = ((p >= 4) ? 16 : 0) + half * 8 + 2 * (p & 3);
      ah.u[p] = *(const unsigned*)&sAhi[arow][ka];
      al.u[p] = *(const unsigned*)&sAlo[arow][ka];
      const int kb = ((p >= 4) ? 16 : 0) + half * 8 + 2 * (p & 3);
      bh0.u[p] = *(const unsigned*)&sBhi[bcol][kb];
      bl0.u[p] = *(const unsigned*)&sBlo[bcol][kb];
      bh1.u[p] = *(const unsigned*)&sBhi[bcol + 16][kb];
      bl1.u[p] = *(const unsigned*)&sBlo[bcol + 16][kb];
    }
    c0 = wmma_bf16(ah.v, bh0.v, c0);
    c0 = wmma_bf16(ah.v, bl0.v, c0);
    c0 = wmma_bf16(al.v, bh0.v, c0);
    c1 = wmma_bf16(ah.v, bh1.v, c1);
    c1 = wmma_bf16(ah.v, bl1.v, c1);
    c1 = wmma_bf16(al.v, bh1.v, c1);
    __syncthreads();
  }
  const float bv0 = bias[n0 + wn * 32 + ln];
  const float bv1 = bias[n0 + wn * 32 + 16 + ln];
#pragma unroll
  for (int v = 0; v < 8; v++) {
    const size_t row = m0 + wm * 16 + v + 8 * half;
    Out[row * (size_t)N + n0 + wn * 32 + ln]      = c0[v] + bv0;
    Out[row * (size_t)N + n0 + wn * 32 + 16 + ln] = c1[v] + bv1;
  }
}

// ---------------------------------------------------------------------------
// In-place RoPE on buf[B*T, H, 128]; thread handles one (row, head, d<64) pair.
// ---------------------------------------------------------------------------
__global__ void rope_kernel(float* __restrict__ buf, int H, size_t total) {
  size_t tid = (size_t)blockIdx.x * blockDim.x + threadIdx.x;
  if (tid >= total) return;
  const int d     = (int)(tid & 63);
  const size_t rh = tid >> 6;  // row*H + h
  const size_t row = rh / (size_t)H;
  const float t = (float)(row & (T_SEQ - 1));
  const float inv = expf(-(float)d * (9.210340371976184f / 64.f));
  float s, cth;
  sincosf(t * inv, &s, &cth);
  const size_t base = rh * DHEAD + d;
  const float x1 = buf[base], x2 = buf[base + 64];
  buf[base]      = x1 * cth - x2 * s;
  buf[base + 64] = x1 * s + x2 * cth;
}

// ---------------------------------------------------------------------------
// Pre-split K (row-major [4096,512]) into bf16 hi/lo, once per element.
// ---------------------------------------------------------------------------
__global__ void pack_split_kernel(const float* __restrict__ src,
                                  __bf16* __restrict__ hi,
                                  __bf16* __restrict__ lo) {
  const size_t t = (size_t)blockIdx.x * blockDim.x + threadIdx.x;  // pair idx
  const float2 v = ((const float2*)src)[t];
  const size_t o = t * 2;
  __bf16 xh = (__bf16)v.x, yh = (__bf16)v.y;
  hi[o] = xh;
  hi[o + 1] = yh;
  lo[o] = (__bf16)(v.x - (float)xh);
  lo[o + 1] = (__bf16)(v.y - (float)yh);
}

// ---------------------------------------------------------------------------
// Pre-split V transposed to [b, g, d, t] bf16 hi/lo so the P@V B-fragment's
// K-dim (key index) pairs are contiguous 4B loads.
// ---------------------------------------------------------------------------
__global__ void pack_vt_kernel(const float* __restrict__ V,
                               __bf16* __restrict__ hi,
                               __bf16* __restrict__ lo) {
  const size_t o2 = (size_t)blockIdx.x * blockDim.x + threadIdx.x;  // pair idx
  const int tp = (int)(o2 & 1023);
  const size_t rest = o2 >> 10;
  const int d  = (int)(rest & 127);
  const int bg = (int)(rest >> 7);
  const int b = bg >> 2, g = bg & 3;
  const int t0 = tp * 2;
  const size_t s0 = ((size_t)(b * T_SEQ + t0) * NGRP + g) * DHEAD + d;
  const float x = V[s0];
  const float y = V[s0 + (size_t)NGRP * DHEAD];
  const size_t o = o2 * 2;
  __bf16 xh = (__bf16)x, yh = (__bf16)y;
  hi[o] = xh;
  hi[o + 1] = yh;
  lo[o] = (__bf16)(x - (float)xh);
  lo[o + 1] = (__bf16)(y - (float)yh);
}

// ---------------------------------------------------------------------------
// Sliding-window + sink flash attention.
// grid: (T/64, B*NHEAD). Block 128 thr = 4 waves; each wave owns 16 queries.
// ---------------------------------------------------------------------------
__global__ void __launch_bounds__(128)
attn_kernel(const float* __restrict__ Q,
            const __bf16* __restrict__ Khi, const __bf16* __restrict__ Klo,
            const __bf16* __restrict__ Vthi, const __bf16* __restrict__ Vtlo,
            float* __restrict__ O) {
  const int bh = blockIdx.y;
  const int b = bh >> 4, hh = bh & 15, g = hh >> 2;
  const int wave = threadIdx.x >> 5, lane = threadIdx.x & 31;
  const int half = lane >> 4, ln = lane & 15;
  const int i0 = blockIdx.x * 64 + wave * 16;
  const size_t qstride = (size_t)NHEAD * DHEAD;  // 2048
  const size_t kstride = (size_t)NGRP * DHEAD;   // 512
  const float* Qp = Q + ((size_t)b * T_SEQ * NHEAD + hh) * DHEAD;
  const __bf16* Khp = Khi + ((size_t)b * T_SEQ * NGRP + g) * DHEAD;
  const __bf16* Klp = Klo + ((size_t)b * T_SEQ * NGRP + g) * DHEAD;
  const __bf16* Vhp = Vthi + (size_t)(b * NGRP + g) * DHEAD * T_SEQ;
  const __bf16* Vlp = Vtlo + (size_t)(b * NGRP + g) * DHEAD * T_SEQ;
  float* Op = O + ((size_t)b * T_SEQ * NHEAD + hh) * DHEAD;

  __shared__ float ptile[4][16][36];  // wave-private C-layout -> A-layout bounce
  float (*pt)[36] = ptile[wave];

  const v8f zero8 = {0.f, 0.f, 0.f, 0.f, 0.f, 0.f, 0.f, 0.f};

  // Q fragments (split once; pre-scaled by 1/sqrt(128))
  v16bf qhi[4], qlo[4];
  {
    const float sc = 0.08838834764831845f;
    const float* qr = Qp + (size_t)(i0 + ln) * qstride;
#pragma unroll
    for (int cc = 0; cc < 4; cc++) {
#pragma unroll
      for (int p = 0; p < 8; p++) {
        const int k = cc * 32 + ((p >= 4) ? 16 : 0) + half * 8 + 2 * (p & 3);
        float2 q2 = *(const float2*)(qr + k);
        split_pair(q2.x * sc, q2.y * sc, qhi[cc], qlo[cc], 2 * p);
      }
    }
  }

  v8f acc[8];
#pragma unroll
  for (int i = 0; i < 8; i++) acc[i] = zero8;
  float mrow[8], lrow[8];
#pragma unroll
  for (int v = 0; v < 8; v++) {
    mrow[v] = -1e30f;
    lrow[v] = 0.f;
  }

  auto tile = [&](int j0) {
    // ---- scores: S[16q x 32k] = Q @ K^T, two 16x16 C tiles ----
    v8f s1 = zero8, s2 = zero8;
    const int rA = imin(j0 + ln, T_SEQ - 1);  // key rows (clamped)
    const int rB = imin(j0 + 16 + ln, T_SEQ - 1);
#pragma unroll
    for (int cc = 0; cc < 4; cc++) {
      Frag kAh, kAl, kBh, kBl;
#pragma unroll
      for (int p = 0; p < 8; p++) {
        const int kk = cc * 32 + ((p >= 4) ? 16 : 0) + half * 8 + 2 * (p & 3);
        kAh.u[p] = *(const unsigned*)&Khp[(size_t)rA * kstride + kk];
        kAl.u[p] = *(const unsigned*)&Klp[(size_t)rA * kstride + kk];
        kBh.u[p] = *(const unsigned*)&Khp[(size_t)rB * kstride + kk];
        kBl.u[p] = *(const unsigned*)&Klp[(size_t)rB * kstride + kk];
      }
      s1 = wmma_bf16(qhi[cc], kAh.v, s1);
      s1 = wmma_bf16(qhi[cc], kAl.v, s1);
      s1 = wmma_bf16(qlo[cc], kAh.v, s1);
      s2 = wmma_bf16(qhi[cc], kBh.v, s2);
      s2 = wmma_bf16(qhi[cc], kBl.v, s2);
      s2 = wmma_bf16(qlo[cc], kBh.v, s2);
    }
    // ---- mask + online softmax (C layout: row = v + 8*half, col = ln) ----
#pragma unroll
    for (int v = 0; v < 8; v++) {
      const int i = i0 + v + 8 * half;
      const int jA = j0 + ln, jB = jA + 16;
      const bool okA = (jA <= i) && ((jA + WIN >= i) || (jA < SINK));
      const bool okB = (jB <= i) && ((jB + WIN >= i) || (jB < SINK));
      s1[v] = okA ? s1[v] : -1e30f;
      s2[v] = okB ? s2[v] : -1e30f;
      float t = fmaxf(s1[v], s2[v]);
      t = fmaxf(t, __shfl_xor(t, 1, 32));
      t = fmaxf(t, __shfl_xor(t, 2, 32));
      t = fmaxf(t, __shfl_xor(t, 4, 32));
      t = fmaxf(t, __shfl_xor(t, 8, 32));
      const float mn = fmaxf(mrow[v], t);
      const float rescale = __expf(mrow[v] - mn);
      mrow[v] = mn;
      s1[v] = __expf(s1[v] - mn);
      s2[v] = __expf(s2[v] - mn);
      float rs = s1[v] + s2[v];
      rs += __shfl_xor(rs, 1, 32);
      rs += __shfl_xor(rs, 2, 32);
      rs += __shfl_xor(rs, 4, 32);
      rs += __shfl_xor(rs, 8, 32);
      lrow[v] = lrow[v] * rescale + rs;
#pragma unroll
      for (int dt = 0; dt < 8; dt++) acc[dt][v] *= rescale;
    }
    // ---- redistribute P from C layout to A layout via wave-private LDS ----
#pragma unroll
    for (int v = 0; v < 8; v++) {
      pt[v + 8 * half][ln] = s1[v];
      pt[v + 8 * half][16 + ln] = s2[v];
    }
    asm volatile("s_wait_dscnt 0" ::: "memory");
    v16bf phi, plo;
#pragma unroll
    for (int p = 0; p < 8; p++) {
      const int k = ((p >= 4) ? 16 : 0) + half * 8 + 2 * (p & 3);
      float2 p2 = *(const float2*)&pt[ln][k];
      split_pair(p2.x, p2.y, phi, plo, 2 * p);
    }
    // ---- acc[16q x 128d] += P[16x32] @ V[32 x 128] (8 d-tiles) ----
#pragma unroll
    for (int dt = 0; dt < 8; dt++) {
      const __bf16* vrh = Vhp + (size_t)(dt * 16 + ln) * T_SEQ;
      const __bf16* vrl = Vlp + (size_t)(dt * 16 + ln) * T_SEQ;
      Frag vh, vl;
#pragma unroll
      for (int p = 0; p < 8; p++) {
        const int idx = j0 + ((p >= 4) ? 16 : 0) + half * 8 + 2 * (p & 3);
        vh.u[p] = *(const unsigned*)&vrh[idx];
        vl.u[p] = *(const unsigned*)&vrl[idx];
      }
      acc[dt] = wmma_bf16(phi, vh.v, acc[dt]);
      acc[dt] = wmma_bf16(phi, vl.v, acc[dt]);
      acc[dt] = wmma_bf16(plo, vh.v, acc[dt]);
    }
  };

  int jstart = i0 - WIN;
  jstart = (jstart < 0) ? 0 : (jstart & ~31);
  if (jstart > 0) tile(0);  // attention-sink tile (mask keeps only j<4)
  for (int j0 = jstart; j0 <= i0 + 15; j0 += 32) tile(j0);

  float inv[8];
#pragma unroll
  for (int v = 0; v < 8; v++) inv[v] = 1.f / lrow[v];
#pragma unroll
  for (int dt = 0; dt < 8; dt++) {
#pragma unroll
    for (int v = 0; v < 8; v++) {
      Op[(size_t)(i0 + v + 8 * half) * qstride + dt * 16 + ln] = acc[dt][v] * inv[v];
    }
  }
}

// ---------------------------------------------------------------------------
extern "C" void kernel_launch(void* const* d_in, const int* in_sizes, int n_in,
                              void* d_out, int out_size, void* d_ws, size_t ws_size,
                              hipStream_t stream) {
  const float* x  = (const float*)d_in[0];
  const float* Wq = (const float*)d_in[1];
  const float* bq = (const float*)d_in[2];
  const float* Wk = (const float*)d_in[3];
  const float* bk = (const float*)d_in[4];
  const float* Wv = (const float*)d_in[5];
  const float* bv = (const float*)d_in[6];
  const float* Wo = (const float*)d_in[7];
  const float* bo = (const float*)d_in[8];
  float* out = (float*)d_out;

  const size_t QN = (size_t)4096 * 2048;  // Q / attn-out elements
  const size_t KN = (size_t)4096 * 512;   // K / V elements
  float* Qb = (float*)d_ws;
  float* Kb = Qb + QN;
  float* Vb = Kb + KN;
  float* Ab = Vb + KN;
  __bf16* Khi  = (__bf16*)(Ab + QN);
  __bf16* Klo  = Khi + KN;
  __bf16* Vthi = Klo + KN;
  __bf16* Vtlo = Vthi + KN + 64;  // 64-elem slack for masked tail reads

  // QKV projections (fp32-accurate split-bf16 WMMA GEMM, async LDS staging)
  gemm_bias_kernel<<<dim3(64, 32), 256, 0, stream>>>(x, Wq, bq, Qb, 4096, 2048, 2048);
  gemm_bias_kernel<<<dim3(64, 8), 256, 0, stream>>>(x, Wk, bk, Kb, 4096, 512, 2048);
  gemm_bias_kernel<<<dim3(64, 8), 256, 0, stream>>>(x, Wv, bv, Vb, 4096, 512, 2048);

  // RoPE on Q and K (in place)
  rope_kernel<<<(4096u * 16 * 64) / 256, 256, 0, stream>>>(Qb, 16, (size_t)4096 * 16 * 64);
  rope_kernel<<<(4096u * 4 * 64) / 256, 256, 0, stream>>>(Kb, 4, (size_t)4096 * 4 * 64);

  // Pre-split K (row-major) and V (transposed) into bf16 hi/lo
  pack_split_kernel<<<(unsigned)(KN / 2 / 256), 256, 0, stream>>>(Kb, Khi, Klo);
  pack_vt_kernel<<<(unsigned)(KN / 2 / 256), 256, 0, stream>>>(Vb, Vthi, Vtlo);

  // Sliding-window flash attention: grid (T/64, B*NHEAD)
  attn_kernel<<<dim3(32, 32), 128, 0, stream>>>(Qb, Khi, Klo, Vthi, Vtlo, Ab);

  // Output projection
  gemm_bias_kernel<<<dim3(64, 32), 256, 0, stream>>>(Ab, Wo, bo, out, 4096, 2048, 2048);
}
